// VQLayer_31748398252207
// MI455X (gfx1250) — matrix operand order repaired and checked by reading, not compiled, not run on hardware
//
#include <hip/hip_runtime.h>

// Problem constants (from reference)
#define N_TOTAL 32768
#define DIM 64
#define K_CODES 1024

#define THREADS 256
#define ROWS_PER_WG 128                    // 8 waves * 16 rows
#define NUM_WG (N_TOTAL / ROWS_PER_WG)     // 256
#define NUM_CT (K_CODES / 16)              // 64 code tiles of 16
#define CT_PER_GROUP 8                     // 128 codes staged per LDS group
#define NUM_GROUPS (NUM_CT / CT_PER_GROUP) // 8
#define GROUP_V16 (CT_PER_GROUP * 2 * 32)  // 512 v16bf fragments per group buffer
#define COPIES (GROUP_V16 / THREADS)       // 2 v16bf per thread per buffer

typedef __attribute__((ext_vector_type(16))) __bf16 v16bf;
typedef __attribute__((ext_vector_type(8)))  float  v8f;

// Workspace layout (bytes):
//   [0, 128K)    : fragHi   64 ct * 2 dc * 32 lanes * 16 bf16 (B-matrix WMMA layout)
//   [128K, 256K) : fragLo
//   [256K, 260K) : eSq      1024 f32
#define FRAG_ELEMS   (NUM_CT * 2 * 32 * 16)   // 65536 bf16
#define LO_OFF_BYTES (FRAG_ELEMS * 2)         // 131072
#define ESQ_OFF_BYTES (2 * LO_OFF_BYTES)      // 262144

// ---------------------------------------------------------------------------
// CDNA5 async copy: global -> LDS, 16B per lane, tracked by ASYNCcnt.
//   vdst = per-lane LDS byte offset, vaddr = per-lane byte offset, saddr = base
// ---------------------------------------------------------------------------
__device__ __forceinline__ void async_copy_b128(unsigned int lds_off,
                                                unsigned int voff,
                                                unsigned long long sbase) {
  asm volatile("global_load_async_to_lds_b128 %0, %1, %2"
               :: "v"(lds_off), "v"(voff), "s"(sbase)
               : "memory");
}
__device__ __forceinline__ void wait_async0() {
  asm volatile("s_wait_asynccnt 0x0" ::: "memory");
}

// ---------------------------------------------------------------------------
// Prep: split codebook into bf16 hi/lo B-fragments, pre-swizzled for
// v_wmma_f32_16x16x32_bf16 (B 32x16: lane&15 = column n, j -> K = 16*(lane>>4)+j),
// and compute e_sq. 4096 threads, one (ct,dc,lane) slot each.
// ---------------------------------------------------------------------------
__global__ void __launch_bounds__(THREADS)
vq_prep(const float* __restrict__ emb, __bf16* __restrict__ fragHi,
        __bf16* __restrict__ fragLo, float* __restrict__ eSq) {
  const int t    = blockIdx.x * THREADS + threadIdx.x;  // 0..4095
  const int ct   = t >> 6;
  const int dc   = (t >> 5) & 1;
  const int lane = t & 31;
  const int code  = ct * 16 + (lane & 15);
  const int dbase = dc * 32 + 16 * (lane >> 4);
  const float* src = emb + code * DIM + dbase;          // 16 contiguous f32
  v16bf hi, lo;
#pragma unroll
  for (int j = 0; j < 16; ++j) {
    float x = src[j];
    __bf16 h = (__bf16)x;                 // RNE
    hi[j] = h;
    lo[j] = (__bf16)(x - (float)h);       // residual for bf16x2 precision
  }
  const int fi = (ct * 2 + dc) * 32 + lane;
  ((v16bf*)fragHi)[fi] = hi;
  ((v16bf*)fragLo)[fi] = lo;

  if (t < K_CODES) {                      // ||e||^2 per code
    const float* er = emb + t * DIM;
    float s = 0.f;
#pragma unroll
    for (int d = 0; d < DIM; ++d) s += er[d] * er[d];
    eSq[t] = s;
  }
}

// ---------------------------------------------------------------------------
// Main: per-wave 16-row tile vs all 1024 codes via bf16x2 WMMA, running
// argmax of (cross - 0.5*e_sq)  ==  argmin of reference distance.
// Fragment groups double-buffered in LDS via global_load_async_to_lds_b128.
// ---------------------------------------------------------------------------
__global__ void __launch_bounds__(THREADS)
vq_main(const float* __restrict__ inp, const float* __restrict__ emb,
        const __bf16* __restrict__ fragHi, const __bf16* __restrict__ fragLo,
        const float* __restrict__ eSq, float* __restrict__ out) {
  __shared__ __bf16 sHi[2][GROUP_V16 * 16];   // 2 x 16 KB
  __shared__ __bf16 sLo[2][GROUP_V16 * 16];   // 2 x 16 KB
  __shared__ float  sEsq[K_CODES];            // 4 KB
  __shared__ int    sBest[ROWS_PER_WG];       // 0.5 KB

  const int tid  = threadIdx.x;
  const int lane = tid & 31;
  const int wave = tid >> 5;
  const int half = lane >> 4;   // 0 | 1
  const int l15  = lane & 15;

  // LDS byte offsets of the double buffers (low 32 bits of flat LDS address)
  const unsigned int ldsHi0 = (unsigned int)(uintptr_t)&sHi[0][0];
  const unsigned int ldsHi1 = (unsigned int)(uintptr_t)&sHi[1][0];
  const unsigned int ldsLo0 = (unsigned int)(uintptr_t)&sLo[0][0];
  const unsigned int ldsLo1 = (unsigned int)(uintptr_t)&sLo[1][0];
  const unsigned long long gHiAddr = (unsigned long long)(uintptr_t)fragHi;
  const unsigned long long gLoAddr = (unsigned long long)(uintptr_t)fragLo;

  // --- stage group 0 into parity buffer 0 (async) ---
  {
    const unsigned int gByte = 0u;
#pragma unroll
    for (int i = 0; i < COPIES; ++i) {
      const unsigned int eoff = (unsigned int)((tid + i * THREADS) * 32);
#pragma unroll
      for (int q = 0; q < 2; ++q) {
        async_copy_b128(ldsHi0 + eoff + q * 16, gByte + eoff + q * 16, gHiAddr);
        async_copy_b128(ldsLo0 + eoff + q * 16, gByte + eoff + q * 16, gLoAddr);
      }
    }
  }

  // --- e_sq into LDS (regular path, overlaps with async) ---
#pragma unroll
  for (int i = 0; i < K_CODES / THREADS; ++i) sEsq[tid + i * THREADS] = eSq[tid + i * THREADS];

  // --- A fragments (16x32 bf16, layout: lane<16 -> K in {0..15}, lane>=16 -> K in {16..31}) ---
  const int row = blockIdx.x * ROWS_PER_WG + wave * 16 + l15;
  const float* rp = inp + row * DIM;
  v16bf aHi[2], aLo[2];
#pragma unroll
  for (int dc = 0; dc < 2; ++dc) {
    const int off = dc * 32 + 16 * half;
    float av[16];
    float4 p0 = *(const float4*)(rp + off);
    float4 p1 = *(const float4*)(rp + off + 4);
    float4 q0 = *(const float4*)(rp + off + 8);
    float4 q1 = *(const float4*)(rp + off + 12);
    av[0]=p0.x; av[1]=p0.y; av[2]=p0.z; av[3]=p0.w;
    av[4]=p1.x; av[5]=p1.y; av[6]=p1.z; av[7]=p1.w;
    av[8]=q0.x; av[9]=q0.y; av[10]=q0.z; av[11]=q0.w;
    av[12]=q1.x; av[13]=q1.y; av[14]=q1.z; av[15]=q1.w;
#pragma unroll
    for (int j = 0; j < 16; ++j) {
      __bf16 h = (__bf16)av[j];
      aHi[dc][j] = h;
      aLo[dc][j] = (__bf16)(av[j] - (float)h);
    }
  }

  float best[8];
  int   bidx[8];
#pragma unroll
  for (int r = 0; r < 8; ++r) { best[r] = -3.4e38f; bidx[r] = 0; }

  wait_async0();
  __syncthreads();      // group 0 resident in buffer 0

  for (int g = 0; g < NUM_GROUPS; ++g) {
    const int p = g & 1;

    // Kick off next group into the other buffer; overlaps with WMMA below.
    if (g + 1 < NUM_GROUPS) {
      const unsigned int gByte = (unsigned int)((g + 1) * GROUP_V16 * 32);
      const unsigned int dHi = p ? ldsHi0 : ldsHi1;
      const unsigned int dLo = p ? ldsLo0 : ldsLo1;
#pragma unroll
      for (int i = 0; i < COPIES; ++i) {
        const unsigned int eoff = (unsigned int)((tid + i * THREADS) * 32);
#pragma unroll
        for (int q = 0; q < 2; ++q) {
          async_copy_b128(dHi + eoff + q * 16, gByte + eoff + q * 16, gHiAddr);
          async_copy_b128(dLo + eoff + q * 16, gByte + eoff + q * 16, gLoAddr);
        }
      }
    }

    const v16bf* lHi = (const v16bf*)&sHi[p][0];
    const v16bf* lLo = (const v16bf*)&sLo[p][0];

    for (int c = 0; c < CT_PER_GROUP; ++c) {
      const int ct = g * CT_PER_GROUP + c;
      const float eq = sEsq[ct * 16 + l15];            // ds_load, column-constant
      v8f acc;
#pragma unroll
      for (int r = 0; r < 8; ++r) acc[r] = -0.5f * eq; // score = cross - 0.5*e_sq
#pragma unroll
      for (int dc = 0; dc < 2; ++dc) {
        const int bi = (c * 2 + dc) * 32 + lane;
        v16bf bHi = lHi[bi];
        v16bf bLo = lLo[bi];
        acc = __builtin_amdgcn_wmma_f32_16x16x32_bf16(false, aHi[dc], false, bHi,
                                                      (short)0, acc, false, false);
        acc = __builtin_amdgcn_wmma_f32_16x16x32_bf16(false, aHi[dc], false, bLo,
                                                      (short)0, acc, false, false);
        acc = __builtin_amdgcn_wmma_f32_16x16x32_bf16(false, aLo[dc], false, bHi,
                                                      (short)0, acc, false, false);
      }
      const int code = ct * 16 + l15;
#pragma unroll
      for (int r = 0; r < 8; ++r) {
        if (acc[r] > best[r]) { best[r] = acc[r]; bidx[r] = code; }  // keep first on tie
      }
    }

    wait_async0();      // next group's async copies landed (this wave's)
    __syncthreads();    // all waves done reading p / writing p^1
  }

  // Reduce over the 16 columns held in each 16-lane half (C layout:
  // lanes 0-15 -> rows r, lanes 16-31 -> rows r+8). argmax, lower idx on tie.
#pragma unroll
  for (int r = 0; r < 8; ++r) {
#pragma unroll
    for (int m = 8; m >= 1; m >>= 1) {   // stays within each 16-lane half
      float ov = __shfl_xor(best[r], m, 32);
      int   oi = __shfl_xor(bidx[r], m, 32);
      if (ov > best[r] || (ov == best[r] && oi < bidx[r])) { best[r] = ov; bidx[r] = oi; }
    }
  }
  if (l15 == 0) {
#pragma unroll
    for (int r = 0; r < 8; ++r) sBest[wave * 16 + half * 8 + r] = bidx[r];
  }
  __syncthreads();

  // Epilogue: quantized = closest + (x - closest), f32 exactly as reference.
  const int base = blockIdx.x * ROWS_PER_WG * DIM;
  for (int e = tid; e < ROWS_PER_WG * DIM; e += THREADS) {
    const int rrow = e >> 6;
    const int col  = e & 63;
    const float x = inp[base + e];
    const float cc = emb[sBest[rrow] * DIM + col];
    out[base + e] = cc + (x - cc);
  }
}

// ---------------------------------------------------------------------------
extern "C" void kernel_launch(void* const* d_in, const int* in_sizes, int n_in,
                              void* d_out, int out_size, void* d_ws, size_t ws_size,
                              hipStream_t stream) {
  (void)in_sizes; (void)n_in; (void)out_size; (void)ws_size;
  const float* inp = (const float*)d_in[0];   // [32,32,32,64] f32
  const float* emb = (const float*)d_in[1];   // [1024,64] f32
  float* out = (float*)d_out;

  __bf16* fragHi = (__bf16*)d_ws;
  __bf16* fragLo = (__bf16*)((char*)d_ws + LO_OFF_BYTES);
  float*  eSqBuf = (float*)((char*)d_ws + ESQ_OFF_BYTES);

  vq_prep<<<(NUM_CT * 2 * 32) / THREADS, THREADS, 0, stream>>>(emb, fragHi, fragLo, eSqBuf);
  vq_main<<<NUM_WG, THREADS, 0, stream>>>(inp, emb, fragHi, fragLo, eSqBuf, out);
}
